// KeypointRelationHead_63642825392170
// MI455X (gfx1250) — matrix operand order repaired and checked by reading, not compiled, not run on hardware
//
#include <hip/hip_runtime.h>
#include <hip/hip_bf16.h>

#define J_K 17
#define C48 48
#define C96 96
#define HM 28
#define HW 784          // 28*28
#define IN_HW 3136      // 56*56
#define NT 49           // 784 / 16 tiles
#define NB 16

typedef _Float16 h8  __attribute__((ext_vector_type(8)));
typedef _Float16 v16h __attribute__((ext_vector_type(16)));
typedef float    v8f  __attribute__((ext_vector_type(8)));

// ---------------------------------------------------------------------------
// Kernel 1: stride-2 3x3 conv + relu, producing stacked features in f16.
//   c96 in [0,48):  vis path (IC=48, W_vis),  scale = W_pred[c96]       (w_v)
//   c96 in [48,96): cls path (IC=18, W_cls),  scale = W_pred[c96]       (w_c)
// Output layout (fragment friendly, c contiguous):
//   Ft[(b*784 + i)*96 + c96] = relu(conv)          (B-side, unscaled)
//   At[(b*784 + i)*96 + c96] = relu(conv)*scale    (A-side, scaled)
// ---------------------------------------------------------------------------
__global__ void conv_feat_kernel(const float* __restrict__ vis_maps,
                                 const float* __restrict__ kpt,
                                 const float* __restrict__ W_vis,
                                 const float* __restrict__ W_cls,
                                 const float* __restrict__ W_pred,
                                 _Float16* __restrict__ At,
                                 _Float16* __restrict__ Ft) {
    __shared__ float s_in[IN_HW];     // one 56x56 input channel
    __shared__ float s_w[C48 * 9];    // weights for this output channel

    const int c96 = blockIdx.x;       // 0..95
    const int b   = blockIdx.y;       // 0..15
    const int tid = threadIdx.x;      // 0..255

    const bool is_vis = (c96 < C48);
    const int  oc = is_vis ? c96 : (c96 - C48);
    const int  IC = is_vis ? C48 : (J_K + 1);
    const float scale = W_pred[c96];

    // stage weights for this oc: IC*9 floats
    const float* wsrc = is_vis ? (W_vis + oc * C48 * 9) : (W_cls + oc * (J_K + 1) * 9);
    for (int idx = tid; idx < IC * 9; idx += blockDim.x) s_w[idx] = wsrc[idx];

    float acc0 = 0.f, acc1 = 0.f, acc2 = 0.f, acc3 = 0.f;

    for (int ic = 0; ic < IC; ++ic) {
        const float* src = is_vis ? (vis_maps + ((size_t)(b * C48 + ic)) * IN_HW)
                                  : (kpt      + ((size_t)(b * (J_K + 1) + ic)) * IN_HW);
        __syncthreads();
        for (int idx = tid; idx < IN_HW; idx += blockDim.x) s_in[idx] = src[idx];
        __syncthreads();

        const float* w = s_w + ic * 9;
#pragma unroll
        for (int o = 0; o < 4; ++o) {
            const int i = tid + o * 256;
            if (i >= HW) break;
            const int oy = i / HM, ox = i % HM;
            float a = 0.f;
#pragma unroll
            for (int ky = 0; ky < 3; ++ky) {
                const int iy = oy * 2 - 1 + ky;
                if (iy < 0 || iy > 55) continue;
#pragma unroll
                for (int kx = 0; kx < 3; ++kx) {
                    const int ix = ox * 2 - 1 + kx;
                    if (ix < 0 || ix > 55) continue;
                    a = fmaf(w[ky * 3 + kx], s_in[iy * 56 + ix], a);
                }
            }
            if (o == 0) acc0 += a; else if (o == 1) acc1 += a;
            else if (o == 2) acc2 += a; else acc3 += a;
        }
    }

#pragma unroll
    for (int o = 0; o < 4; ++o) {
        const int i = tid + o * 256;
        if (i >= HW) break;
        float v = (o == 0) ? acc0 : (o == 1) ? acc1 : (o == 2) ? acc2 : acc3;
        v = fmaxf(v, 0.f);
        const size_t base = ((size_t)(b * HW + i)) * C96 + c96;
        Ft[base] = (_Float16)v;
        At[base] = (_Float16)(v * scale);
    }
}

// ---------------------------------------------------------------------------
// Kernel 2: a[c,i] = sum_f W_loc[c,f] * loc_maps[f, i, 0]
// (loc_maps[f,i,j] = lm[f,i]-lm[f,j], so pairwise diffs of a[] reproduce geo)
// ---------------------------------------------------------------------------
__global__ void loc_proj_kernel(const float* __restrict__ W_loc,
                                const float* __restrict__ loc_maps,
                                float* __restrict__ a_buf) {
    const int idx = blockIdx.x * blockDim.x + threadIdx.x;
    if (idx >= C48 * HW) return;
    const int c = idx / HW, i = idx % HW;
    float s = 0.f;
#pragma unroll
    for (int f = 0; f < 4; ++f)
        s = fmaf(W_loc[c * 4 + f], loc_maps[(size_t)f * HW * HW + (size_t)i * HW], s);
    a_buf[idx] = s;
}

// ---------------------------------------------------------------------------
// Kernel 3: geo[i,j] = sum_c w_g[c] * relu(a[c,i] - a[c,j])   (batch-invariant)
// 16x16 tile per block; a-slices staged in LDS.
// ---------------------------------------------------------------------------
__global__ void geo_tile_kernel(const float* __restrict__ a_buf,
                                const float* __restrict__ W_pred,
                                float* __restrict__ geo) {
    __shared__ float sI[C48 * 16];
    __shared__ float sJ[C48 * 16];
    __shared__ float swg[C48];

    const int i0 = blockIdx.x * 16, j0 = blockIdx.y * 16;
    const int tid = threadIdx.x;

    for (int idx = tid; idx < C48 * 16; idx += blockDim.x) {
        const int c = idx >> 4, t = idx & 15;
        sI[idx] = a_buf[c * HW + i0 + t];
        sJ[idx] = a_buf[c * HW + j0 + t];
    }
    if (tid < C48) swg[tid] = W_pred[2 * C48 + tid];
    __syncthreads();

    const int ri = tid >> 4, cj = tid & 15;
    float g = 0.f;
#pragma unroll 8
    for (int c = 0; c < C48; ++c)
        g = fmaf(swg[c], fmaxf(sI[c * 16 + ri] - sJ[c * 16 + cj], 0.f), g);
    geo[(size_t)(i0 + ri) * HW + j0 + cj] = g;
}

// ---------------------------------------------------------------------------
// Kernel 4: per-batch Gram matrix via WMMA, K=96 (3 x 16x16x32 f16 WMMA),
// + geo add + fast sigmoid. One wave per 16x16 output tile. 784 = 49*16.
//
// Fragment gathers per documented CDNA5 VGPR striping (hi = lane>>4, m = lane&15):
//   A (16x32 f16): elem e -> K = e + (e>=8 ? 8 : 0) + hi*8, M = m
//       => two contiguous b128 loads at row(i0+m): [kb+hi*8, +8) and [kb+16+hi*8, +8)
//   B (32x16 f16): elem e -> K = e + (e>=8 ? 8 : 0) + hi*8, N = m
//       => two contiguous b128 loads at row(j0+m): [kb+hi*8, +8) and [kb+16+hi*8, +8)
//   C/D (16x16 f32): VGPR r -> M = r + hi*8, N = m
// ---------------------------------------------------------------------------
__device__ __forceinline__ v16h load_a_frag(const _Float16* __restrict__ row,
                                            int kb, int hi) {
    union { v16h v; h8 h[2]; } u;
    u.h[0] = *reinterpret_cast<const h8*>(row + kb + hi * 8);
    u.h[1] = *reinterpret_cast<const h8*>(row + kb + 16 + hi * 8);
    return u.v;
}

__device__ __forceinline__ v16h load_b_frag(const _Float16* __restrict__ row,
                                            int kb, int hi) {
    union { v16h v; h8 h[2]; } u;
    u.h[0] = *reinterpret_cast<const h8*>(row + kb + hi * 8);
    u.h[1] = *reinterpret_cast<const h8*>(row + kb + 16 + hi * 8);
    return u.v;
}

__global__ void gram_wmma_kernel(const _Float16* __restrict__ At,
                                 const _Float16* __restrict__ Ft,
                                 const float* __restrict__ geo,
                                 float* __restrict__ out) {
    const int lane = threadIdx.x & 31;
    const int wave = threadIdx.x >> 5;          // 0..6
    const int m  = lane & 15;
    const int hi = lane >> 4;

    const int i0 = blockIdx.x * 16;                      // row tile
    const int j0 = (blockIdx.y * 7 + wave) * 16;         // col tile (49 total)
    const int b  = blockIdx.z;

    const _Float16* rowA = At + ((size_t)(b * HW + i0 + m)) * C96;
    const _Float16* rowB = Ft + ((size_t)(b * HW + j0 + m)) * C96;

    v8f acc = {};
#pragma unroll
    for (int kb = 0; kb < C96; kb += 32) {
        v16h a = load_a_frag(rowA, kb, hi);
        v16h bm = load_b_frag(rowB, kb, hi);
        acc = __builtin_amdgcn_wmma_f32_16x16x32_f16(
            /*neg_a=*/false, a, /*neg_b=*/false, bm,
            /*c_mod=*/(short)0, acc, /*reuse_a=*/false, /*reuse_b=*/false);
    }

    // Epilogue: geo add + fast sigmoid. Use raw v_rcp_f32 instead of the IEEE
    // division sequence (div_scale/div_fmas/div_fixup) -- this kernel must be
    // store-bound, not VALU-bound, and the logits are tiny (~1e-3).
    float* obase = out + (size_t)b * HW * HW;
#pragma unroll
    for (int r = 0; r < 8; ++r) {
        const int Mrow = i0 + r + hi * 8;
        const int Ncol = j0 + m;
        const float x = acc[r] + geo[(size_t)Mrow * HW + Ncol];
        const float e = __expf(-x);                       // v_exp_f32
        obase[(size_t)Mrow * HW + Ncol] = __builtin_amdgcn_rcpf(1.0f + e); // v_rcp_f32
    }
}

// ---------------------------------------------------------------------------
// Host-side launcher
// ---------------------------------------------------------------------------
extern "C" void kernel_launch(void* const* d_in, const int* in_sizes, int n_in,
                              void* d_out, int out_size, void* d_ws, size_t ws_size,
                              hipStream_t stream) {
    const float* vis_maps = (const float*)d_in[0];   // (16,48,56,56)
    const float* kpt      = (const float*)d_in[1];   // (16,18,56,56)
    const float* W_cls    = (const float*)d_in[2];   // (48,18,3,3)
    const float* W_vis    = (const float*)d_in[3];   // (48,48,3,3)
    const float* W_loc    = (const float*)d_in[4];   // (48,4)
    const float* W_pred   = (const float*)d_in[5];   // (144,)
    const float* loc_maps = (const float*)d_in[6];   // (4,784,784)
    float* out = (float*)d_out;                      // (16,784,784)

    // workspace layout
    char* ws = (char*)d_ws;
    const size_t featBytes = (size_t)NB * HW * C96 * sizeof(_Float16);   // 2,408,448
    _Float16* At   = (_Float16*)ws;
    _Float16* Ft   = (_Float16*)(ws + featBytes);
    float*    aBuf = (float*)(ws + 2 * featBytes);
    float*    geo  = (float*)(ws + 2 * featBytes + (size_t)C48 * HW * sizeof(float));

    // 1) convolutions -> stacked f16 features (scaled + raw)
    conv_feat_kernel<<<dim3(C96, NB), 256, 0, stream>>>(
        vis_maps, kpt, W_vis, W_cls, W_pred, At, Ft);

    // 2) a[c,i] projection (48*784 = 37,632 elements)
    loc_proj_kernel<<<(C48 * HW + 255) / 256, 256, 0, stream>>>(
        W_loc, loc_maps, aBuf);

    // 3) geo[i,j] (batch-invariant, 2.4 MB -> lives in L2 for reuse)
    geo_tile_kernel<<<dim3(NT, NT), 256, 0, stream>>>(aBuf, W_pred, geo);

    // 4) WMMA Gram + geo + sigmoid (7 waves/block, one 16x16 tile per wave)
    gram_wmma_kernel<<<dim3(NT, 7, NB), 224, 0, stream>>>(At, Ft, geo, out);
}
